// SelfAttention_19765439496481
// MI455X (gfx1250) — matrix operand order repaired and checked by reading, not compiled, not run on hardware
//
#include <hip/hip_runtime.h>
#include <hip/hip_bf16.h>
#include <stdint.h>

// ---------------------------------------------------------------------------
// Self-attention for MI455X (gfx1250):
//   - all matmuls via v_wmma_f32_16x16x32_f16
//   - LDS staging via global_load_async_to_lds_b128 (ASYNCcnt), double-buffered
// ---------------------------------------------------------------------------

typedef __attribute__((ext_vector_type(16))) _Float16 v16h;
typedef __attribute__((ext_vector_type(8)))  _Float16 v8h;
typedef __attribute__((ext_vector_type(4)))  _Float16 v4h;
typedef __attribute__((ext_vector_type(8)))  float    v8f;
typedef __attribute__((ext_vector_type(4)))  float    v4f;

static constexpr int kB   = 4;       // batch
static constexpr int kS   = 2048;    // sequence
static constexpr int kD   = 1024;    // d_model
static constexpr int kH   = 16;      // heads
static constexpr int kDH  = 64;      // head dim
static constexpr int kN3  = 3 * kD;  // 3072
static constexpr int kM   = kB * kS; // 8192 tokens

__device__ __forceinline__ v8f zero_v8f() {
    v8f z;
    #pragma unroll
    for (int i = 0; i < 8; ++i) z[i] = 0.0f;
    return z;
}

__device__ __forceinline__ v16h cat16(v8h lo, v8h hi) {
    return __builtin_shufflevector(lo, hi, 0,1,2,3,4,5,6,7,8,9,10,11,12,13,14,15);
}

__device__ __forceinline__ v8f wmma_f16(v16h a, v16h b, v8f c) {
    return __builtin_amdgcn_wmma_f32_16x16x32_f16(false, a, false, b, (short)0, c, false, false);
}

// Async 16-byte global -> LDS copy (per lane), tracked by ASYNCcnt.
__device__ __forceinline__ void async_cp16(void* lds, const _Float16* g) {
    uint32_t l = (uint32_t)(uintptr_t)lds;
    asm volatile("global_load_async_to_lds_b128 %0, %1, off"
                 :: "v"(l), "v"(g) : "memory");
}
__device__ __forceinline__ void wait_async0() {
    asm volatile("s_wait_asynccnt 0x0" ::: "memory");
}

// A-operand per-lane k pattern: lo = k in [8h, 8h+7], hi = k in [16+8h, 23+8h]
// B-operand per-lane k pattern: same as A: lo = k in [8h, 8h+7], hi = k in [16+8h, 23+8h]

// ---------------------------------------------------------------------------
// Kernel 0a: fp32 -> f16 elementwise (for x)
// ---------------------------------------------------------------------------
__global__ __launch_bounds__(256) void cvt_f16_kernel(
    const float* __restrict__ in, _Float16* __restrict__ out)
{
    const size_t i = (size_t)(blockIdx.x * 256 + threadIdx.x) * 4;
    v4f v = *(const v4f*)&in[i];
    v4h o;
    #pragma unroll
    for (int j = 0; j < 4; ++j) o[j] = (_Float16)v[j];
    *(v4h*)&out[i] = o;
}

// ---------------------------------------------------------------------------
// Kernel 0b: W [K][N] fp32 -> WT [N][K] f16 (tiled transpose + convert)
// ---------------------------------------------------------------------------
__global__ __launch_bounds__(256) void transpose_cvt_kernel(
    const float* __restrict__ W, _Float16* __restrict__ WT, int K, int N)
{
    __shared__ __attribute__((aligned(16))) _Float16 t[32][40];
    const int kb = blockIdx.y * 32, nb = blockIdx.x * 32;
    const int r = threadIdx.x >> 3, c4 = threadIdx.x & 7;
    v4f v = *(const v4f*)&W[(size_t)(kb + r) * N + nb + c4 * 4];
    #pragma unroll
    for (int j = 0; j < 4; ++j) t[c4 * 4 + j][r] = (_Float16)v[j];
    __syncthreads();
    const int cc = (threadIdx.x & 7) * 4;
    v4h o;
    #pragma unroll
    for (int j = 0; j < 4; ++j) o[j] = t[r][cc + j];
    *(v4h*)&WT[(size_t)(nb + r) * K + kb + cc] = o;
}

// ---------------------------------------------------------------------------
// Kernel 1: qkv = x16 @ WqkvT^T + b  ->  scatter to Q (scaled), K, V^T (f16)
//   M=8192, N=3072, K=1024.  WG tile 128x128, 8 waves of 32x64.
//   Async double-buffered LDS staging.
// ---------------------------------------------------------------------------
__global__ __launch_bounds__(256) void qkv_gemm_kernel(
    const _Float16* __restrict__ x16, const _Float16* __restrict__ WT,
    const float* __restrict__ bqkv,
    _Float16* __restrict__ Qb, _Float16* __restrict__ Kb,
    _Float16* __restrict__ VTb)
{
    __shared__ __attribute__((aligned(16))) _Float16 ldsA[2][128][40]; // [m][k]
    __shared__ __attribute__((aligned(16))) _Float16 ldsB[2][128][40]; // [n][k]

    const int tid   = threadIdx.x;
    const int Mbase = blockIdx.y * 128;
    const int Nbase = blockIdx.x * 128;
    const int w  = tid >> 5, lane = tid & 31;
    const int wm = w & 3,    wn   = w >> 2;
    const int ml = lane & 15, hl  = lane >> 4;

    v8f acc[2][4];
    #pragma unroll
    for (int t = 0; t < 2; ++t)
        #pragma unroll
        for (int u = 0; u < 4; ++u) acc[t][u] = zero_v8f();

    // per-thread chunk coords: A/B tiles are 128 rows x 4 x (8 f16) = 512 chunks
    const int cr0 = tid >> 2, cch = tid & 3;        // chunk set 0
    const int cr1 = (tid + 256) >> 2;               // chunk set 1 (same cch)

    auto issue = [&](int buf, int kbase) {
        async_cp16(&ldsA[buf][cr0][cch * 8], &x16[(size_t)(Mbase + cr0) * kD + kbase + cch * 8]);
        async_cp16(&ldsA[buf][cr1][cch * 8], &x16[(size_t)(Mbase + cr1) * kD + kbase + cch * 8]);
        async_cp16(&ldsB[buf][cr0][cch * 8], &WT[(size_t)(Nbase + cr0) * kD + kbase + cch * 8]);
        async_cp16(&ldsB[buf][cr1][cch * 8], &WT[(size_t)(Nbase + cr1) * kD + kbase + cch * 8]);
    };

    issue(0, 0);
    const int KSTEPS = kD / 32;
    for (int kb = 0; kb < KSTEPS; ++kb) {
        wait_async0();
        __syncthreads();
        const int cur = kb & 1;
        if (kb + 1 < KSTEPS) issue(cur ^ 1, (kb + 1) * 32);

        v16h afr[2], bfr[4];
        #pragma unroll
        for (int t = 0; t < 2; ++t) {
            int row = wm * 32 + t * 16 + ml;
            v8h lo = *(const v8h*)&ldsA[cur][row][hl * 8];
            v8h hi = *(const v8h*)&ldsA[cur][row][16 + hl * 8];
            afr[t] = cat16(lo, hi);
        }
        #pragma unroll
        for (int u = 0; u < 4; ++u) {
            int col = wn * 64 + u * 16 + ml;
            v8h lo = *(const v8h*)&ldsB[cur][col][hl * 8];
            v8h hi = *(const v8h*)&ldsB[cur][col][16 + hl * 8];
            bfr[u] = cat16(lo, hi);
        }
        #pragma unroll
        for (int t = 0; t < 2; ++t)
            #pragma unroll
            for (int u = 0; u < 4; ++u)
                acc[t][u] = wmma_f16(afr[t], bfr[u], acc[t][u]);
    }

    // epilogue: bias + route columns to Q/K/V^T
    const float scale = 0.125f; // 1/sqrt(64)
    #pragma unroll
    for (int t = 0; t < 2; ++t) {
        #pragma unroll
        for (int u = 0; u < 4; ++u) {
            const int n    = Nbase + wn * 64 + u * 16 + ml;
            const int head = n / 192, rsel = n % 192;
            const float bias = bqkv[n];
            #pragma unroll
            for (int r = 0; r < 8; ++r) {
                const int m = Mbase + wm * 32 + t * 16 + r + 8 * hl;
                const float val = acc[t][u][r] + bias;
                const int b = m >> 11, s = m & (kS - 1);
                const size_t bh = (size_t)(b * kH + head);
                if (rsel < 64) {
                    Qb[(bh * kS + s) * kDH + rsel] = (_Float16)(val * scale);
                } else if (rsel < 128) {
                    Kb[(bh * kS + s) * kDH + (rsel - 64)] = (_Float16)val;
                } else {
                    VTb[(bh * kDH + (rsel - 128)) * kS + s] = (_Float16)val;
                }
            }
        }
    }
}

// ---------------------------------------------------------------------------
// Kernel 2: flash attention per (b,h). WG = 8 waves x 32 query rows.
//   Async double-buffered K / V^T tiles of 32 keys.
// ---------------------------------------------------------------------------
__global__ __launch_bounds__(256) void attn_kernel(
    const _Float16* __restrict__ Qb, const _Float16* __restrict__ Kb,
    const _Float16* __restrict__ VTb, _Float16* __restrict__ attn16)
{
    __shared__ __attribute__((aligned(16))) _Float16 ldsK[2][32][72];    // [key][dh]
    __shared__ __attribute__((aligned(16))) _Float16 ldsVT[2][64][40];   // [dh][key]
    __shared__ __attribute__((aligned(16))) _Float16 ldsP[8][32][40];    // per-wave P

    const int tid  = threadIdx.x;
    const int bh   = blockIdx.y;
    const int b    = bh >> 4, h = bh & 15;
    const int w    = tid >> 5, lane = tid & 31;
    const int ml   = lane & 15, hl = lane >> 4;
    const int qbase = blockIdx.x * 256 + w * 32;

    const size_t qkoff = (size_t)bh * kS * kDH;
    const size_t vtoff = (size_t)bh * kDH * kS;

    // Q fragments (scale pre-folded), whole dh=64, held in registers
    v16h qf[2][2];
    #pragma unroll
    for (int t = 0; t < 2; ++t) {
        const _Float16* qrow = Qb + qkoff + (size_t)(qbase + t * 16 + ml) * kDH;
        #pragma unroll
        for (int kd = 0; kd < 2; ++kd) {
            v8h lo = *(const v8h*)&qrow[kd * 32 + hl * 8];
            v8h hi = *(const v8h*)&qrow[kd * 32 + 16 + hl * 8];
            qf[t][kd] = cat16(lo, hi);
        }
    }

    float mrun[2][8], lrun[2][8];
    v8f Oacc[2][4];
    #pragma unroll
    for (int t = 0; t < 2; ++t) {
        #pragma unroll
        for (int r = 0; r < 8; ++r) { mrun[t][r] = -3.0e38f; lrun[t][r] = 0.0f; }
        #pragma unroll
        for (int u = 0; u < 4; ++u) Oacc[t][u] = zero_v8f();
    }

    // per-thread async-copy coords: K tile 32x64 (256 chunks), VT 64x32 (256)
    const int krow = tid >> 3, kch = tid & 7;
    const int vrow = tid >> 2, vch = tid & 3;

    auto issue = [&](int buf, int keybase) {
        async_cp16(&ldsK[buf][krow][kch * 8],
                   &Kb[qkoff + (size_t)(keybase + krow) * kDH + kch * 8]);
        async_cp16(&ldsVT[buf][vrow][vch * 8],
                   &VTb[vtoff + (size_t)vrow * kS + keybase + vch * 8]);
    };

    issue(0, 0);
    const int NBLK = kS / 32;
    for (int kb = 0; kb < NBLK; ++kb) {
        wait_async0();
        __syncthreads();
        const int cur = kb & 1;
        if (kb + 1 < NBLK) issue(cur ^ 1, (kb + 1) * 32);

        // scores: 32 q rows x 32 keys
        v8f sc[2][2];
        #pragma unroll
        for (int t = 0; t < 2; ++t) {
            #pragma unroll
            for (int j = 0; j < 2; ++j) {
                v8f s = zero_v8f();
                #pragma unroll
                for (int kd = 0; kd < 2; ++kd) {
                    int key = j * 16 + ml;
                    v8h lo = *(const v8h*)&ldsK[cur][key][kd * 32 + hl * 8];
                    v8h hi = *(const v8h*)&ldsK[cur][key][kd * 32 + 16 + hl * 8];
                    s = wmma_f16(qf[t][kd], cat16(lo, hi), s);
                }
                sc[t][j] = s;
            }
        }

        // online softmax (row stats across 16-lane halves)
        #pragma unroll
        for (int t = 0; t < 2; ++t) {
            float alpha[8];
            #pragma unroll
            for (int r = 0; r < 8; ++r) {
                float mx = fmaxf(sc[t][0][r], sc[t][1][r]);
                #pragma unroll
                for (int off = 8; off >= 1; off >>= 1)
                    mx = fmaxf(mx, __shfl_xor(mx, off, 32));
                float mnew = fmaxf(mrun[t][r], mx);
                float a = __expf(mrun[t][r] - mnew);
                mrun[t][r] = mnew;
                float p0 = __expf(sc[t][0][r] - mnew);
                float p1 = __expf(sc[t][1][r] - mnew);
                sc[t][0][r] = p0; sc[t][1][r] = p1;
                float ps = p0 + p1;
                #pragma unroll
                for (int off = 8; off >= 1; off >>= 1)
                    ps += __shfl_xor(ps, off, 32);
                lrun[t][r] = lrun[t][r] * a + ps;
                alpha[r] = a;
            }
            #pragma unroll
            for (int u = 0; u < 4; ++u)
                #pragma unroll
                for (int r = 0; r < 8; ++r) Oacc[t][u][r] *= alpha[r];
            #pragma unroll
            for (int j = 0; j < 2; ++j)
                #pragma unroll
                for (int r = 0; r < 8; ++r)
                    ldsP[w][t * 16 + r + 8 * hl][j * 16 + ml] = (_Float16)sc[t][j][r];
        }

        // O += P @ V
        v16h bV[4];
        #pragma unroll
        for (int u = 0; u < 4; ++u) {
            v8h lo = *(const v8h*)&ldsVT[cur][u * 16 + ml][hl * 8];
            v8h hi = *(const v8h*)&ldsVT[cur][u * 16 + ml][16 + hl * 8];
            bV[u] = cat16(lo, hi);
        }
        #pragma unroll
        for (int t = 0; t < 2; ++t) {
            v8h lo = *(const v8h*)&ldsP[w][t * 16 + ml][hl * 8];
            v8h hi = *(const v8h*)&ldsP[w][t * 16 + ml][16 + hl * 8];
            v16h aP = cat16(lo, hi);
            #pragma unroll
            for (int u = 0; u < 4; ++u) Oacc[t][u] = wmma_f16(aP, bV[u], Oacc[t][u]);
        }
    }

    // finalize: O / l -> attn16 token-major [B,S,H*dh]
    #pragma unroll
    for (int t = 0; t < 2; ++t) {
        #pragma unroll
        for (int r = 0; r < 8; ++r) {
            const int m = qbase + t * 16 + r + 8 * hl;
            const float inv = 1.0f / lrun[t][r];
            const size_t base = ((size_t)b * kS + m) * kD + h * kDH;
            #pragma unroll
            for (int u = 0; u < 4; ++u)
                attn16[base + u * 16 + ml] = (_Float16)(Oacc[t][u][r] * inv);
        }
    }
}

// ---------------------------------------------------------------------------
// Kernel 3: out = attn16 @ WoutT^T + b_out (M=8192, N=1024, K=1024), fp32 out
// ---------------------------------------------------------------------------
__global__ __launch_bounds__(256) void out_gemm_kernel(
    const _Float16* __restrict__ A, const _Float16* __restrict__ WT,
    const float* __restrict__ bout, float* __restrict__ out)
{
    __shared__ __attribute__((aligned(16))) _Float16 ldsA[2][128][40];
    __shared__ __attribute__((aligned(16))) _Float16 ldsB[2][128][40];

    const int tid   = threadIdx.x;
    const int Mbase = blockIdx.y * 128;
    const int Nbase = blockIdx.x * 128;
    const int w  = tid >> 5, lane = tid & 31;
    const int wm = w & 3,    wn   = w >> 2;
    const int ml = lane & 15, hl  = lane >> 4;

    v8f acc[2][4];
    #pragma unroll
    for (int t = 0; t < 2; ++t)
        #pragma unroll
        for (int u = 0; u < 4; ++u) acc[t][u] = zero_v8f();

    const int cr0 = tid >> 2, cch = tid & 3;
    const int cr1 = (tid + 256) >> 2;

    auto issue = [&](int buf, int kbase) {
        async_cp16(&ldsA[buf][cr0][cch * 8], &A[(size_t)(Mbase + cr0) * kD + kbase + cch * 8]);
        async_cp16(&ldsA[buf][cr1][cch * 8], &A[(size_t)(Mbase + cr1) * kD + kbase + cch * 8]);
        async_cp16(&ldsB[buf][cr0][cch * 8], &WT[(size_t)(Nbase + cr0) * kD + kbase + cch * 8]);
        async_cp16(&ldsB[buf][cr1][cch * 8], &WT[(size_t)(Nbase + cr1) * kD + kbase + cch * 8]);
    };

    issue(0, 0);
    const int KSTEPS = kD / 32;
    for (int kb = 0; kb < KSTEPS; ++kb) {
        wait_async0();
        __syncthreads();
        const int cur = kb & 1;
        if (kb + 1 < KSTEPS) issue(cur ^ 1, (kb + 1) * 32);

        v16h afr[2], bfr[4];
        #pragma unroll
        for (int t = 0; t < 2; ++t) {
            int row = wm * 32 + t * 16 + ml;
            v8h lo = *(const v8h*)&ldsA[cur][row][hl * 8];
            v8h hi = *(const v8h*)&ldsA[cur][row][16 + hl * 8];
            afr[t] = cat16(lo, hi);
        }
        #pragma unroll
        for (int u = 0; u < 4; ++u) {
            int col = wn * 64 + u * 16 + ml;
            v8h lo = *(const v8h*)&ldsB[cur][col][hl * 8];
            v8h hi = *(const v8h*)&ldsB[cur][col][16 + hl * 8];
            bfr[u] = cat16(lo, hi);
        }
        #pragma unroll
        for (int t = 0; t < 2; ++t)
            #pragma unroll
            for (int u = 0; u < 4; ++u)
                acc[t][u] = wmma_f16(afr[t], bfr[u], acc[t][u]);
    }

    #pragma unroll
    for (int t = 0; t < 2; ++t) {
        #pragma unroll
        for (int u = 0; u < 4; ++u) {
            const int n = Nbase + wn * 64 + u * 16 + ml;
            const float bias = bout[n];
            #pragma unroll
            for (int r = 0; r < 8; ++r) {
                const int m = Mbase + wm * 32 + t * 16 + r + 8 * hl;
                out[(size_t)m * kD + n] = acc[t][u][r] + bias;
            }
        }
    }
}

// ---------------------------------------------------------------------------
extern "C" void kernel_launch(void* const* d_in, const int* in_sizes, int n_in,
                              void* d_out, int out_size, void* d_ws, size_t ws_size,
                              hipStream_t stream) {
    (void)in_sizes; (void)n_in; (void)out_size; (void)ws_size;

    const float* x    = (const float*)d_in[0];
    const float* Wqkv = (const float*)d_in[1];
    const float* bqkv = (const float*)d_in[2];
    const float* Wout = (const float*)d_in[3];
    const float* bout = (const float*)d_in[4];
    float* out = (float*)d_out;

    char* ws = (char*)d_ws;
    const size_t segQ = (size_t)kB * kH * kS * kDH * sizeof(_Float16); // 16 MB
    _Float16* Qb     = (_Float16*)(ws + 0 * segQ);
    _Float16* Kb     = (_Float16*)(ws + 1 * segQ);
    _Float16* VTb    = (_Float16*)(ws + 2 * segQ);
    _Float16* attn16 = (_Float16*)(ws + 3 * segQ);
    char* p = ws + 4 * segQ;
    _Float16* x16    = (_Float16*)p;                 p += (size_t)kM * kD * 2;   // 16 MB
    _Float16* WqkvT  = (_Float16*)p;                 p += (size_t)kN3 * kD * 2;  // 6 MB
    _Float16* WoutT  = (_Float16*)p;                                             // 2 MB

    // precision prep (bandwidth-trivial): x->f16, W->f16 transposed
    cvt_f16_kernel<<<(kM * kD / 4) / 256, 256, 0, stream>>>(x, x16);
    transpose_cvt_kernel<<<dim3(kN3 / 32, kD / 32), 256, 0, stream>>>(Wqkv, WqkvT, kD, kN3);
    transpose_cvt_kernel<<<dim3(kD / 32, kD / 32), 256, 0, stream>>>(Wout, WoutT, kD, kD);

    qkv_gemm_kernel<<<dim3(kN3 / 128, kM / 128), 256, 0, stream>>>(
        x16, WqkvT, bqkv, Qb, Kb, VTb);
    attn_kernel<<<dim3(kS / 256, kB * kH), 256, 0, stream>>>(
        Qb, Kb, VTb, attn16);
    out_gemm_kernel<<<dim3(kD / 128, kM / 128), 256, 0, stream>>>(
        attn16, WoutT, bout, out);
}
